// boundary_dice_bce_loss_28870770164377
// MI455X (gfx1250) — compile-verified, matrix operand order checked by reading, not run on hardware
//
#include <hip/hip_runtime.h>
#include <hip/hip_bf16.h>

typedef float v2f  __attribute__((ext_vector_type(2)));
typedef float v8f  __attribute__((ext_vector_type(8)));
typedef _Float16 v16h __attribute__((ext_vector_type(16)));
typedef unsigned long long u64;

#define BATCH 16
#define HH 1024
#define WW 1024
#define NELEM (BATCH * HH * WW)      // 16,777,216
#define N4 (NELEM / 4)               // 4,194,304 float4s
#define WPR 16                       // u64 words per row (1024 bits)
#define WPI (HH * WPR)               // 16384 words per image
#define NWORDS (BATCH * WPI)         // 262,144 words per mask
#define RAD 29                       // dilation radius

#define K1_BLOCKS 1024
#define K1_THREADS 256

// ---------------------------------------------------------------------------
// Wave-level sum of 64 f32 values (2 per lane) via WMMA with all-ones B:
// D[m][n] = sum_k A[m][k] + C[m][n]  ->  total = sum over any column of D.
// Lane layout of A is irrelevant for a *total* sum.
// ---------------------------------------------------------------------------
__device__ __forceinline__ float wave_sum64_wmma(v2f acc) {
#if __has_builtin(__builtin_amdgcn_wmma_f32_16x16x4_f32)
  v2f bones = {1.0f, 1.0f};
  v8f c = {};
  c = __builtin_amdgcn_wmma_f32_16x16x4_f32(
      false, acc, false, bones, (short)0, c, false, false);
  float s = c[0] + c[1] + c[2] + c[3] + c[4] + c[5] + c[6] + c[7];
  return s + __shfl_xor(s, 16, 32);
#elif __has_builtin(__builtin_amdgcn_wmma_f32_16x16x32_f16)
  v16h a = {};
  a[0] = (_Float16)acc.x;
  a[1] = (_Float16)acc.y;
  v16h bh;
#pragma unroll
  for (int i = 0; i < 16; ++i) bh[i] = (_Float16)1.0f;
  v8f c = {};
  c = __builtin_amdgcn_wmma_f32_16x16x32_f16(
      false, a, false, bh, (short)0, c, false, false);
  float s = c[0] + c[1] + c[2] + c[3] + c[4] + c[5] + c[6] + c[7];
  return s + __shfl_xor(s, 16, 32);
#else
  float s = acc.x + acc.y;
#pragma unroll
  for (int d = 16; d > 0; d >>= 1) s += __shfl_xor(s, d, 32);
  return s;
#endif
}

// ---------------------------------------------------------------------------
// K1: single pass over the 128MB of f32 input (the only HBM-bound kernel).
//  - binarize -> bit-packed masks (32x compression; masks live in L2 after)
//  - accumulate sum(y_true), sum(y_pred), sum(y_true*y_pred) per lane,
//    reduced per-wave via WMMA, per-block via LDS -> deterministic partials.
// ---------------------------------------------------------------------------
__global__ __launch_bounds__(K1_THREADS)
void k1_bin_sums(const float* __restrict__ yt, const float* __restrict__ yp,
                 u64* __restrict__ yt_bits, u64* __restrict__ yp_bits,
                 float* __restrict__ partials, u64* __restrict__ counters) {
  if (blockIdx.x == 0 && threadIdx.x == 0) {
    counters[0] = 0ull; counters[1] = 0ull; counters[2] = 0ull;
  }
  __shared__ unsigned char nib_t[K1_THREADS];
  __shared__ unsigned char nib_p[K1_THREADS];
  __shared__ float wpart[3][K1_THREADS / 32];

  const float4* yt4 = (const float4*)yt;
  const float4* yp4 = (const float4*)yp;
  const unsigned tid = threadIdx.x;

  v2f at = {0.0f, 0.0f}, ap = {0.0f, 0.0f}, ai = {0.0f, 0.0f};

  // N4 % (K1_BLOCKS*K1_THREADS) == 0 -> uniform trip count, barriers are safe.
  for (unsigned idx4 = blockIdx.x * K1_THREADS + tid; idx4 < N4;
       idx4 += K1_BLOCKS * K1_THREADS) {
    float4 t = yt4[idx4];
    float4 p = yp4[idx4];

    at.x += t.x + t.z;            at.y += t.y + t.w;
    ap.x += p.x + p.z;            ap.y += p.y + p.w;
    ai.x += t.x * p.x + t.z * p.z; ai.y += t.y * p.y + t.w * p.w;

    unsigned nt = (t.x >= 0.5f ? 1u : 0u) | (t.y >= 0.5f ? 2u : 0u) |
                  (t.z >= 0.5f ? 4u : 0u) | (t.w >= 0.5f ? 8u : 0u);
    unsigned np = (p.x >= 0.5f ? 1u : 0u) | (p.y >= 0.5f ? 2u : 0u) |
                  (p.z >= 0.5f ? 4u : 0u) | (p.w >= 0.5f ? 8u : 0u);

    __syncthreads();  // previous iteration's readers are done
    nib_t[tid] = (unsigned char)nt;
    nib_p[tid] = (unsigned char)np;
    __syncthreads();
    if ((tid & 15u) == 0u) {
      u64 wt = 0ull, wp = 0ull;
#pragma unroll
      for (int j = 0; j < 16; ++j) {
        wt |= (u64)(nib_t[tid + j] & 0xFu) << (4 * j);
        wp |= (u64)(nib_p[tid + j] & 0xFu) << (4 * j);
      }
      yt_bits[idx4 >> 4] = wt;   // 64 consecutive pixels -> one word
      yp_bits[idx4 >> 4] = wp;
    }
  }

  float st = wave_sum64_wmma(at);
  float sp = wave_sum64_wmma(ap);
  float si = wave_sum64_wmma(ai);

  const unsigned wave = tid >> 5;
  if ((tid & 31u) == 0u) {
    wpart[0][wave] = st; wpart[1][wave] = sp; wpart[2][wave] = si;
  }
  __syncthreads();
  if (tid == 0) {
    float a = 0.f, b = 0.f, c = 0.f;
#pragma unroll
    for (int w = 0; w < K1_THREADS / 32; ++w) {
      a += wpart[0][w]; b += wpart[1][w]; c += wpart[2][w];
    }
    partials[blockIdx.x * 3 + 0] = a;
    partials[blockIdx.x * 3 + 1] = b;
    partials[blockIdx.x * 3 + 2] = c;
  }
}

// ---------------------------------------------------------------------------
// K2: horizontal 59-tap erosion (min == AND for binary) per bit-packed row.
// Left-aligned erosion via 6 shift-AND doubling steps (1,2,4,8,16,27 -> 59),
// then re-center with a 29-bit left shift. Zero-fill reproduces the zero ring.
// One thread owns one 1024-bit row (16 u64 in registers, fully unrolled).
// ---------------------------------------------------------------------------
template <int S>
__device__ __forceinline__ void erode_step(u64* r) {
#pragma unroll
  for (int j = 0; j < 16; ++j) r[j] &= (r[j] >> S) | (r[j + 1] << (64 - S));
}

__global__ __launch_bounds__(256)
void k2_herode(const u64* __restrict__ yt_bits, const u64* __restrict__ yp_bits,
               u64* __restrict__ ht_bits, u64* __restrict__ hp_bits) {
  unsigned t = blockIdx.x * blockDim.x + threadIdx.x;  // 2*16384 rows
  if (t >= 2u * BATCH * HH) return;
  const unsigned tensor = t >> 14;
  const unsigned rowBase = (t & 16383u) * WPR;
  const u64* src = tensor ? yp_bits : yt_bits;
  u64* dst = tensor ? hp_bits : ht_bits;

  u64 r[17];
#pragma unroll
  for (int j = 0; j < 16; ++j) r[j] = src[rowBase + j];
  r[16] = 0ull;

  erode_step<1>(r);  erode_step<2>(r);  erode_step<4>(r);
  erode_step<8>(r);  erode_step<16>(r); erode_step<27>(r);

  // centered erosion: eroded[p] = LA[p-29]  ->  shift left by 29 bits
#pragma unroll
  for (int j = 15; j >= 1; --j) r[j] = (r[j] << RAD) | (r[j - 1] >> (64 - RAD));
  r[0] <<= RAD;

#pragma unroll
  for (int j = 0; j < 16; ++j) dst[rowBase + j] = r[j];
}

// ---------------------------------------------------------------------------
// K3: vertical 59-tap erosion (AND over rows, zero ring -> rows near the
// border are forced 0), boundary = mask & ~eroded, popcount the three
// boundary sums. Wave-reduce a packed 3x21-bit counter, u64 atomics (exact,
// order-independent -> deterministic). All operands are L2-resident (8 MB).
// ---------------------------------------------------------------------------
__global__ __launch_bounds__(256)
void k3_verode_count(const u64* __restrict__ yt_bits,
                     const u64* __restrict__ yp_bits,
                     const u64* __restrict__ ht, const u64* __restrict__ hp,
                     u64* __restrict__ counters) {
  const unsigned t = blockIdx.x * blockDim.x + threadIdx.x;  // NWORDS threads
  const unsigned rem = t & 16383u;
  const unsigned r = rem >> 4;
  const unsigned wc = rem & 15u;
  const unsigned base = (t >> 14) * WPI + wc;

  u64 et, ep;
  if (r < RAD || r >= HH - RAD) {
    et = 0ull; ep = 0ull;  // zero ring reaches this row
  } else {
    et = ~0ull; ep = ~0ull;
    for (int dr = -RAD; dr <= RAD; ++dr) {
      const unsigned idx = base + (unsigned)(r + dr) * WPR;
      et &= ht[idx];
      ep &= hp[idx];
    }
  }
  const u64 mt = yt_bits[t];
  const u64 mp = yp_bits[t];
  const u64 bt = mt & ~et;
  const u64 bp = mp & ~ep;

  u64 packed = (u64)__popcll(bt) | ((u64)__popcll(bp) << 21) |
               ((u64)__popcll(bt & bp) << 42);
#pragma unroll
  for (int d = 16; d > 0; d >>= 1) packed += __shfl_xor(packed, d, 32);

  if ((threadIdx.x & 31u) == 0u) {
    atomicAdd(&counters[0], packed & 0x1FFFFFull);
    atomicAdd(&counters[1], (packed >> 21) & 0x1FFFFFull);
    atomicAdd(&counters[2], packed >> 42);
  }
}

// ---------------------------------------------------------------------------
// K4: finalize. Serial double-precision sum of the 3072 block partials
// (deterministic), combine both dice terms.
// ---------------------------------------------------------------------------
__global__ void k4_final(const float* __restrict__ partials,
                         const u64* __restrict__ counters,
                         float* __restrict__ out) {
  if (blockIdx.x == 0 && threadIdx.x == 0) {
    double st = 0.0, sp = 0.0, si = 0.0;
    for (int i = 0; i < K1_BLOCKS; ++i) {
      st += (double)partials[i * 3 + 0];
      sp += (double)partials[i * 3 + 1];
      si += (double)partials[i * 3 + 2];
    }
    const double b1 = 1.0 - (2.0 * si + 1.0) / (st + sp + 1.0);
    const double cbt = (double)counters[0];
    const double cbp = (double)counters[1];
    const double cbi = (double)counters[2];
    const double b2 = 1.0 - (2.0 * cbi + 1.0) / (cbt + cbp + 1.0);
    out[0] = (float)(b1 + b2);
  }
}

// ---------------------------------------------------------------------------
extern "C" void kernel_launch(void* const* d_in, const int* in_sizes, int n_in,
                              void* d_out, int out_size, void* d_ws,
                              size_t ws_size, hipStream_t stream) {
  const float* yt = (const float*)d_in[0];
  const float* yp = (const float*)d_in[1];
  float* out = (float*)d_out;

  u64* ws = (u64*)d_ws;
  u64* yt_bits = ws;                 // 262,144 u64 (2 MB)
  u64* yp_bits = ws + NWORDS;        // 2 MB
  u64* ht_bits = ws + 2 * NWORDS;    // 2 MB
  u64* hp_bits = ws + 3 * NWORDS;    // 2 MB
  u64* counters = ws + 4 * NWORDS;   // 3 u64 (pad to 8 for alignment)
  float* partials = (float*)(ws + 4 * NWORDS + 8);  // 1024*3 floats

  k1_bin_sums<<<K1_BLOCKS, K1_THREADS, 0, stream>>>(yt, yp, yt_bits, yp_bits,
                                                    partials, counters);
  k2_herode<<<(2 * BATCH * HH) / 256, 256, 0, stream>>>(yt_bits, yp_bits,
                                                        ht_bits, hp_bits);
  k3_verode_count<<<NWORDS / 256, 256, 0, stream>>>(yt_bits, yp_bits, ht_bits,
                                                    hp_bits, counters);
  k4_final<<<1, 64, 0, stream>>>(partials, counters, out);
}
